// EdgeFFN_37391985279642
// MI455X (gfx1250) — matrix operand order repaired and checked by reading, not compiled, not run on hardware
//
#include <hip/hip_runtime.h>

typedef _Float16 v16h __attribute__((ext_vector_type(16)));
typedef _Float16 v8h  __attribute__((ext_vector_type(8)));
typedef _Float16 v4h  __attribute__((ext_vector_type(4)));
typedef float    v8f  __attribute__((ext_vector_type(8)));

#define LDW     264       // padded f16 row stride (16B-aligned rows, bank-spread)
#define TWO_IN  256       // concat input dim (2*128)
#define HID     256
#define OUTD    64
#define EPB     128       // edges per block (8 waves x 16-edge strips)

// LDS layout (dynamic): W1^T [256][LDW], W2^T [64][LDW], A/H tile [128][LDW]
#define SMEM_HALVES ((256 + 64 + 128) * LDW)
#define SMEM_BYTES  (SMEM_HALVES * 2)

__device__ __forceinline__ v16h cat8(v8h lo, v8h hi) {
  return __builtin_shufflevector(lo, hi, 0,1,2,3,4,5,6,7,8,9,10,11,12,13,14,15);
}

__global__ __launch_bounds__(256, 1)
void edge_ffn_wmma(const float* __restrict__ x,
                   const int*   __restrict__ ei,    // [2][E]
                   const float* __restrict__ W1,    // [256][256] row-major (k, n)
                   const float* __restrict__ b1,    // [256]
                   const float* __restrict__ W2,    // [256][64]  row-major (k, n)
                   const float* __restrict__ b2,    // [64]
                   float*       __restrict__ out,   // [E][64]
                   int E)
{
  extern __shared__ char smem_raw[];
  _Float16* sW1T = (_Float16*)smem_raw;          // [n=0..255][LDW] : W1T[n][k]
  _Float16* sW2T = sW1T + 256 * LDW;             // [n=0..63][LDW]  : W2T[n][k]
  _Float16* sAH  = sW2T + 64  * LDW;             // [row=0..127][LDW] A then H

  const int tid = threadIdx.x;
  const int e0  = blockIdx.x * EPB;

  // ---- stage W1^T (f32 -> f16). thread t = output row n; coalesced global reads.
  {
    const int n = tid;                           // 0..255
    for (int k = 0; k < TWO_IN; ++k)
      sW1T[n * LDW + k] = (_Float16)W1[k * HID + n];
    const int n2 = tid & 63;
    const int kq = tid >> 6;                     // quarter of k-range
    for (int kk = 0; kk < 64; ++kk) {
      const int k = kq * 64 + kk;
      sW2T[n2 * LDW + k] = (_Float16)W2[k * OUTD + n2];
    }
  }

  // ---- gather concat(x[src], x[dst]) -> sAH (f16). thread t: row t/2, half t&1.
  // wave w's threads cover exactly rows [16w, 16w+16) -> strip stays wave-private.
  {
    const int r = tid >> 1;
    const int p = tid & 1;                       // 0 = src half, 1 = dst half
    int e = e0 + r; if (e >= E) e = E - 1;       // clamp; tail stores are guarded
    const int idx = ei[(size_t)p * E + e];
    const float4* xr = (const float4*)(x + (size_t)idx * 128);
    _Float16* dp = sAH + r * LDW + p * 128;
#pragma unroll
    for (int i = 0; i < 32; ++i) {
      const float4 v = xr[i];
      v4h h;
      h[0] = (_Float16)v.x; h[1] = (_Float16)v.y;
      h[2] = (_Float16)v.z; h[3] = (_Float16)v.w;
      *(v4h*)(dp + 4 * i) = h;                   // 8B aligned LDS store
    }
  }
  __syncthreads();                               // only block-wide sync needed

  const int lane   = tid & 31;
  const int w      = tid >> 5;                   // wave id -> edge strip
  const int rowSel = lane & 15;                  // A row / B,C,D column within tile
  const int hiSel  = lane >> 4;                  // lane-group K/M select

  const _Float16* aRow = sAH + (16 * w + rowSel) * LDW;

  // ================= Layer 1: H[16,256] = relu(A[16,256] @ W1 + b1) ============
  // Preload ALL A fragments (8 k-steps x 8 VGPRs) so the strip can be
  // overwritten in-place with H.
  v16h af[8];
#pragma unroll
  for (int k = 0; k < 8; ++k) {
    const _Float16* p = aRow + 32 * k + hiSel * 8;     // ISA A-layout (16b, 16x32)
    af[k] = cat8(*(const v8h*)p, *(const v8h*)(p + 16));
  }

#pragma unroll
  for (int nt = 0; nt < 16; ++nt) {
    const float bias = b1[16 * nt + rowSel];           // C col = rowSel for all VGPRs
    v8f c;
#pragma unroll
    for (int i = 0; i < 8; ++i) c[i] = bias;
    const _Float16* brow = sW1T + (16 * nt + rowSel) * LDW + hiSel * 8;
#pragma unroll
    for (int k = 0; k < 8; ++k) {
      const _Float16* bp = brow + 32 * k;              // ISA B-layout (16b, 32x16)
      const v16h bf = cat8(*(const v8h*)bp, *(const v8h*)(bp + 16));
      c = __builtin_amdgcn_wmma_f32_16x16x32_f16(false, af[k], false, bf,
                                                 (short)0, c, false, false);
    }
    // ReLU + write H tile back into the (wave-private) strip, f16
    _Float16* hp = sAH + (16 * w) * LDW + 16 * nt + rowSel;
#pragma unroll
    for (int r = 0; r < 8; ++r) {
      float v = c[r];
      v = v > 0.0f ? v : 0.0f;
      hp[(r + 8 * hiSel) * LDW] = (_Float16)v;         // row M = r + 8*hiSel
    }
  }

  // ================= Layer 2: O[16,64] = H[16,256] @ W2 + b2 ===================
  v16h hf[8];
#pragma unroll
  for (int k = 0; k < 8; ++k) {
    const _Float16* p = aRow + 32 * k + hiSel * 8;     // same strip, now holds H
    hf[k] = cat8(*(const v8h*)p, *(const v8h*)(p + 16));
  }

#pragma unroll
  for (int nt = 0; nt < 4; ++nt) {
    const float bias = b2[16 * nt + rowSel];
    v8f c;
#pragma unroll
    for (int i = 0; i < 8; ++i) c[i] = bias;
    const _Float16* brow = sW2T + (16 * nt + rowSel) * LDW + hiSel * 8;
#pragma unroll
    for (int k = 0; k < 8; ++k) {
      const _Float16* bp = brow + 32 * k;
      const v16h bf = cat8(*(const v8h*)bp, *(const v8h*)(bp + 16));
      c = __builtin_amdgcn_wmma_f32_16x16x32_f16(false, hf[k], false, bf,
                                                 (short)0, c, false, false);
    }
    const int n = 16 * nt + rowSel;
#pragma unroll
    for (int r = 0; r < 8; ++r) {
      const int e = e0 + 16 * w + r + 8 * hiSel;
      if (e < E) out[(size_t)e * OUTD + n] = c[r];     // lanes 0-15: 64B contiguous
    }
  }
}

extern "C" void kernel_launch(void* const* d_in, const int* in_sizes, int n_in,
                              void* d_out, int out_size, void* d_ws, size_t ws_size,
                              hipStream_t stream) {
  (void)n_in; (void)d_ws; (void)ws_size; (void)out_size;
  const float* x  = (const float*)d_in[0];
  const int*   ei = (const int*)  d_in[1];
  const float* W1 = (const float*)d_in[2];
  const float* b1 = (const float*)d_in[3];
  const float* W2 = (const float*)d_in[4];
  const float* b2 = (const float*)d_in[5];
  float* out = (float*)d_out;

  const int E = in_sizes[1] / 2;
  const int grid = (E + EPB - 1) / EPB;

  // 231 KB dynamic LDS (> default cap) — idempotent, capture-safe host call
  hipFuncSetAttribute((const void*)edge_ffn_wmma,
                      hipFuncAttributeMaxDynamicSharedMemorySize, SMEM_BYTES);

  edge_ffn_wmma<<<grid, 256, SMEM_BYTES, stream>>>(x, ei, W1, b1, W2, b2, out, E);
}
